// AlignBlock_28097676051138
// MI455X (gfx1250) — matrix operand order repaired and checked by reading, not compiled, not run on hardware
//
#include <hip/hip_runtime.h>

// ---------------------------------------------------------------------------
// AlignBlock: bottleneck conv3x3 (concat w/ ref) + 4x [offset conv3x3 -> WMMA,
// modulated deformable conv -> VALU gather + small matmul].
// Dense convs: implicit GEMM on v_wmma_f32_16x16x32_bf16 (wave32).
// Loop order: outer K-loop (not unrolled) holds ONE live A fragment from LDS;
// inner unrolled tile loop holds TPB accumulators and streams pre-swizzled
// bf16 B fragments from global.  No register spills, A reuse across tiles.
// ---------------------------------------------------------------------------

#define NFRM   8          // burst frames (B*N)
#define FCH    64         // feature channels
#define HH     96
#define WW     96
#define HW     (HH * WW)  // 9216
#define DG     8          // deform groups
#define CG     (FCH / DG) // 8 channels per group
#define KK     9
#define OFFCH  144        // 2*DG*KK
#define MSKCH  72         // DG*KK
#define OMPAD  224        // 216 padded to 14 tiles of 16

typedef __attribute__((ext_vector_type(16))) __bf16 v16bf;
typedef __attribute__((ext_vector_type(8)))  float  v8f;

__device__ __forceinline__ unsigned short f2bf(float f) {
    unsigned int u = __float_as_uint(f);
    unsigned int r = u + 0x7FFFu + ((u >> 16) & 1u);   // round-to-nearest-even
    return (unsigned short)(r >> 16);
}

// ---------------------------------------------------------------------------
// Pack conv weights [CoutReal][Cin][3][3] (f32, OIHW) into bf16 WMMA B-frags.
// Fragment id = (otile*9 + tap)*(Cin/32) + cchunk ; 512 ushorts per fragment,
// element (lane, slot):  n = lane&15, k = (lane>>4)*16 + slot  (B is 32x16).
// ---------------------------------------------------------------------------
__global__ void pack_weights_kernel(const float* __restrict__ w,
                                    unsigned short* __restrict__ dst,
                                    int coutReal, int coutPad, int cin) {
    int total = (coutPad >> 4) * 9 * (cin >> 5) * 512;
    for (int t = blockIdx.x * blockDim.x + threadIdx.x; t < total;
         t += gridDim.x * blockDim.x) {
        int frag = t >> 9;
        int r    = t & 511;
        int lane = r >> 4;
        int slot = r & 15;
        int cchunks = cin >> 5;
        int cchunk  = frag % cchunks;
        int tmp     = frag / cchunks;
        int tap     = tmp % 9;
        int ot      = tmp / 9;
        int half = lane >> 4;
        int nn   = lane & 15;
        int k    = half * 16 + slot;
        int c    = cchunk * 32 + k;
        int o    = ot * 16 + nn;
        float v  = (o < coutReal) ? w[(o * cin + c) * 9 + tap] : 0.0f;
        dst[t] = f2bf(v);
    }
}

// ---------------------------------------------------------------------------
// Implicit-GEMM conv3x3, stride 1, pad 1.  One workgroup = 192 threads =
// 6 waves; each wave owns 16 pixels of one image row.  The 3x98xCIN halo
// patch is staged into LDS as bf16 once.  Per wave: M=16 pixels,
// N = TPB*16 couts (TPB accumulators), K = 9*CIN.
// ---------------------------------------------------------------------------
template<int CIN, bool CONCAT, int TPB>
__global__ void conv3x3_wmma_kernel(const float* __restrict__ src,
                                    const unsigned short* __restrict__ wpk,
                                    float* __restrict__ out,
                                    int coutPad, int tilesTotal) {
    extern __shared__ unsigned short patch[];   // [3][98][CIN]
    const int CC       = CIN / 32;              // channel chunks per tap
    const int TILE_U16 = 9 * CC * 512;          // ushorts per Cout tile

    int splits = tilesTotal / TPB;
    int bid    = blockIdx.x;
    int sp     = bid % splits;
    int tmp    = bid / splits;
    int y      = tmp % HH;
    int n      = tmp / HH;

    // ---- stage input patch (with zero padding at image borders) ----
    const int E = 3 * 98 * CIN;
    for (int e = threadIdx.x; e < E; e += blockDim.x) {
        int c   = e % CIN;
        int col = (e / CIN) % 98;
        int row = e / (CIN * 98);
        int gy  = y + row - 1;
        int gx  = col - 1;
        float v = 0.0f;
        if (gy >= 0 && gy < HH && gx >= 0 && gx < WW) {
            if (CONCAT) {
                v = (c < FCH) ? src[((size_t)(n * FCH + c)) * HW + gy * WW + gx]
                              : src[((size_t)(c - FCH)) * HW + gy * WW + gx];
            } else {
                v = src[((size_t)(n * CIN + c)) * HW + gy * WW + gx];
            }
        }
        patch[e] = f2bf(v);
    }
    __syncthreads();

    // ---- per-wave WMMA GEMM:  one A fragment live, TPB accumulators ----
    int wv   = threadIdx.x >> 5;      // wave 0..5 -> pixel subtile
    int lane = threadIdx.x & 31;
    int half = lane >> 4;
    int ml   = lane & 15;
    int px   = wv * 16 + ml;          // pixel x for A-fragment row M

    // per-lane base into this block's first tile of pre-swizzled weights
    const unsigned short* wbase = wpk + (size_t)(sp * TPB) * TILE_U16
                                      + (lane << 4);

    v8f acc[TPB];
    #pragma unroll
    for (int ti = 0; ti < TPB; ++ti)
        acc[ti] = (v8f){0.f, 0.f, 0.f, 0.f, 0.f, 0.f, 0.f, 0.f};

    #pragma unroll 1
    for (int kk = 0; kk < 9 * CC; ++kk) {
        int tap    = kk / CC;
        int cc     = kk % CC;
        int prow   = tap / 3;                // gy = y + prow - 1
        int pcol   = px + (tap % 3);         // gx = x + dx (halo offset +1)
        int rcbase = (prow * 98 + pcol) * CIN;

        // A: slots 0-15 -> c = cc*32 + half*16 + {0..15}
        v16bf a;
        {
            const uint4* p0 = reinterpret_cast<const uint4*>(
                &patch[rcbase + cc * 32 + half * 16]);
            uint4* ap = reinterpret_cast<uint4*>(&a);
            ap[0] = p0[0];          // 8 bf16
            ap[1] = p0[1];          // next 8 contiguous elements
        }

        // keep the sequential weight stream warm (global_prefetch_b8)
        __builtin_prefetch(wbase + (size_t)(kk + 1) * 512, 0, 1);

        #pragma unroll
        for (int ti = 0; ti < TPB; ++ti) {
            v16bf b;
            const uint4* q = reinterpret_cast<const uint4*>(
                wbase + (size_t)ti * TILE_U16 + (kk << 9));
            uint4* bp = reinterpret_cast<uint4*>(&b);
            bp[0] = q[0];
            bp[1] = q[1];
            acc[ti] = __builtin_amdgcn_wmma_f32_16x16x32_bf16(
                false, a, false, b, (short)0, acc[ti], false, false);
        }
    }

    // ---- store D: lane has cout = nTile*16+ml, 8 contiguous pixels ----
    #pragma unroll
    for (int ti = 0; ti < TPB; ++ti) {
        int ch = (sp * TPB + ti) * 16 + ml;
        float* ob = out + ((size_t)(n * coutPad + ch)) * HW
                        + y * WW + wv * 16 + half * 8;
        *reinterpret_cast<float4*>(ob) =
            make_float4(acc[ti][0], acc[ti][1], acc[ti][2], acc[ti][3]);
        *reinterpret_cast<float4*>(ob + 4) =
            make_float4(acc[ti][4], acc[ti][5], acc[ti][6], acc[ti][7]);
    }
}

// ---------------------------------------------------------------------------
// Modulated deformable conv 3x3 (torchvision semantics), 8 deform groups.
// One block = one (frame, group, 256-pixel chunk).  Group weights in LDS.
// ---------------------------------------------------------------------------
__global__ void deform_conv_kernel(const float* __restrict__ feat,
                                   const float* __restrict__ offmask, // [N][224][HW]
                                   const float* __restrict__ w,       // [64][8][3][3]
                                   const float* __restrict__ bias,    // [64]
                                   float* __restrict__ out) {
    __shared__ float ws[CG * CG * KK];   // [o][c][t]
    __shared__ float bs[CG];

    int bid  = blockIdx.x;
    int pblk = bid % (HW / 256);
    int g    = (bid / (HW / 256)) % DG;
    int n    = bid / ((HW / 256) * DG);

    for (int e = threadIdx.x; e < CG * CG * KK; e += blockDim.x)
        ws[e] = w[(size_t)(g * CG) * (CG * KK) + e];
    if (threadIdx.x < CG) bs[threadIdx.x] = bias[g * CG + threadIdx.x];
    __syncthreads();

    int p = pblk * 256 + threadIdx.x;
    int y = p / WW, x = p % WW;

    float acc[CG];
    #pragma unroll
    for (int o = 0; o < CG; ++o) acc[o] = bs[o];

    const float* fg = feat + (size_t)(n * FCH + g * CG) * HW;
    const float* om = offmask + (size_t)n * OMPAD * HW;

    for (int t = 0; t < KK; ++t) {
        int offc  = (g * KK + t) * 2;
        float dy  = om[(size_t)offc * HW + p];
        float dx  = om[(size_t)(offc + 1) * HW + p];
        float mr  = om[(size_t)(OFFCH + g * KK + t) * HW + p];
        float m   = 1.0f / (1.0f + __expf(-mr));

        float py = (float)y + (float)(t / 3 - 1) + dy;
        float px = (float)x + (float)(t % 3 - 1) + dx;
        float y0f = floorf(py), x0f = floorf(px);
        float ly = py - y0f, lx = px - x0f;
        int y0 = (int)y0f, x0 = (int)x0f;

        float wy[2] = {1.0f - ly, ly};
        float wx[2] = {1.0f - lx, lx};
        float samp[CG] = {0, 0, 0, 0, 0, 0, 0, 0};

        #pragma unroll
        for (int iy = 0; iy < 2; ++iy) {
            int yy = y0 + iy;
            if (yy < 0 || yy >= HH) continue;
            #pragma unroll
            for (int ix = 0; ix < 2; ++ix) {
                int xx = x0 + ix;
                if (xx < 0 || xx >= WW) continue;
                float wgt = wy[iy] * wx[ix];
                int idx = yy * WW + xx;
                #pragma unroll
                for (int c = 0; c < CG; ++c)
                    samp[c] = fmaf(wgt, fg[(size_t)c * HW + idx], samp[c]);
            }
        }
        #pragma unroll
        for (int c = 0; c < CG; ++c) samp[c] *= m;
        #pragma unroll
        for (int o = 0; o < CG; ++o) {
            float s = 0.0f;
            #pragma unroll
            for (int c = 0; c < CG; ++c)
                s = fmaf(ws[(o * CG + c) * KK + t], samp[c], s);
            acc[o] += s;
        }
    }

    #pragma unroll
    for (int o = 0; o < CG; ++o)
        out[(size_t)(n * FCH + g * CG + o) * HW + p] = acc[o];
}

// ---------------------------------------------------------------------------
// Host-side orchestration (graph-capture safe: launches only, on `stream`).
// Input order: 0 burst_feat, 1 bottleneck_w, then per iter i:
//   off_w{i}: 2,5,8,11  def_w{i}: 3,6,9,12  def_b{i}: 4,7,10,13
// ---------------------------------------------------------------------------
extern "C" void kernel_launch(void* const* d_in, const int* in_sizes, int n_in,
                              void* d_out, int out_size, void* d_ws, size_t ws_size,
                              hipStream_t stream) {
    const float* burst   = (const float*)d_in[0];
    const float* bneck_w = (const float*)d_in[1];
    const float* off_w[4] = {(const float*)d_in[2], (const float*)d_in[5],
                             (const float*)d_in[8], (const float*)d_in[11]};
    const float* def_w[4] = {(const float*)d_in[3], (const float*)d_in[6],
                             (const float*)d_in[9], (const float*)d_in[12]};
    const float* def_b[4] = {(const float*)d_in[4], (const float*)d_in[7],
                             (const float*)d_in[10], (const float*)d_in[13]};
    float* outp = (float*)d_out;

    char* ws = (char*)d_ws;
    const size_t FEAT_BYTES = (size_t)NFRM * FCH * HW * sizeof(float);      // 18.9 MB
    const size_t OFFM_BYTES = (size_t)NFRM * OMPAD * HW * sizeof(float);    // 66.1 MB
    const size_t WPKB_BYTES = (size_t)FCH * (2 * FCH) * 9 * 2;              // 147 KB
    const size_t WPKO_BYTES = (size_t)OMPAD * FCH * 9 * 2;                  // 258 KB

    float* featA = (float*)ws;
    float* featB = (float*)(ws + FEAT_BYTES);
    float* offm  = (float*)(ws + 2 * FEAT_BYTES);
    unsigned short* wpk_b = (unsigned short*)(ws + 2 * FEAT_BYTES + OFFM_BYTES);
    unsigned short* wpk_o[4];
    for (int i = 0; i < 4; ++i)
        wpk_o[i] = (unsigned short*)(ws + 2 * FEAT_BYTES + OFFM_BYTES + WPKB_BYTES
                                        + (size_t)i * WPKO_BYTES);

    // ---- pack all conv weights into bf16 WMMA fragments ----
    pack_weights_kernel<<<64, 256, 0, stream>>>(bneck_w, wpk_b, FCH, FCH, 2 * FCH);
    for (int i = 0; i < 4; ++i)
        pack_weights_kernel<<<64, 256, 0, stream>>>(off_w[i], wpk_o[i],
                                                    OFFCH + MSKCH, OMPAD, FCH);

    // ---- bottleneck conv: concat(frame n, frame 0), 128 -> 64 ----
    {
        size_t lds = 3 * 98 * 128 * sizeof(unsigned short);
        // 4 Cout tiles, 2 per block -> 2 splits
        dim3 grid(NFRM * HH * 2);
        conv3x3_wmma_kernel<128, true, 2><<<grid, 192, lds, stream>>>(
            burst, wpk_b, featA, FCH, FCH / 16);
    }

    // ---- 4 deformable alignment stages ----
    float* cur = featA;
    float* nxt = featB;
    size_t lds64 = 3 * 98 * 64 * sizeof(unsigned short);
    for (int i = 0; i < 4; ++i) {
        // 14 Cout tiles, 7 per block -> 2 splits
        dim3 gridc(NFRM * HH * 2);
        conv3x3_wmma_kernel<64, false, 7><<<gridc, 192, lds64, stream>>>(
            cur, wpk_o[i], offm, OMPAD, OMPAD / 16);

        float* dst = (i == 3) ? outp : nxt;
        dim3 gridd(NFRM * DG * (HW / 256));
        deform_conv_kernel<<<gridd, 256, 0, stream>>>(cur, offm, def_w[i],
                                                      def_b[i], dst);
        float* t = cur; cur = nxt; nxt = t;
    }
}
